// KANLayer_13142599926030
// MI455X (gfx1250) — compile-verified, matrix operand order checked by reading, not run on hardware
//
#include <hip/hip_runtime.h>
#include <hip/hip_bf16.h>
#include <stdint.h>

// ---------------------------------------------------------------------------
// KAN layer as a fused feature-expansion GEMM on CDNA5 WMMA.
//   F[b, i*20+f] = f<19 ? basis_f(x[b,i]) : silu(x[b,i])      (f16)
//   Wt[o, i*20+f] = f<19 ? sf[i,o]*cp[i,o,f] : sf[i,o]        (f16, N-major)
//   out[b,o] = sum_k F[b,k] * Wt[o,k]                          (f32 accum)
// M=4096, N=256, K=5120.
// Waves mapped 2x2 over the 64x64 tile: each wave owns 32x32 (4 C tiles).
// LDS panels padded to a 168-half row stride (84 dwords, 20 mod 64 banks)
// so 16-lane fragment loads are bank-conflict-free.
// ---------------------------------------------------------------------------

typedef _Float16 v16h __attribute__((ext_vector_type(16)));
typedef _Float16 v8h  __attribute__((ext_vector_type(8)));
typedef float    v8f  __attribute__((ext_vector_type(8)));

#define IN_DIM   256
#define OUT_DIM  256
#define NUM_CP   19      // 16 intervals + order 3
#define FEAT     20      // 19 basis + silu
#define KDIM     (IN_DIM * FEAT)          // 5120
#define M_TILE   64
#define N_TILE   64
#define I_CHUNK  8                        // inputs per K panel
#define K_PANEL  (I_CHUNK * FEAT)         // 160 = 5 * 32 (data halves per row)
#define K_STRIDE 168                      // padded row stride (bank-friendly)
#define THREADS  128                      // 4 wave32

union Frag { v16h v; v8h h[2]; };

// ---------------------------------------------------------------------------
// Kernel 1: pack weights into Wt[o][i*20+f] as f16.
// ---------------------------------------------------------------------------
__global__ void kan_pack_weights(const float* __restrict__ cp,
                                 const float* __restrict__ sf,
                                 _Float16*    __restrict__ Wt) {
  int idx = blockIdx.x * blockDim.x + threadIdx.x;   // one (i,o) pair
  if (idx >= IN_DIM * OUT_DIM) return;
  int i = idx >> 8;          // / 256
  int o = idx & 255;         // % 256
  float s = sf[i * OUT_DIM + o];
  const float* c = cp + ((size_t)(i * OUT_DIM + o)) * NUM_CP;
  _Float16* w = Wt + (size_t)o * KDIM + i * FEAT;
#pragma unroll
  for (int k = 0; k < NUM_CP; ++k) w[k] = (_Float16)(s * c[k]);
  w[NUM_CP] = (_Float16)s;   // silu channel
}

// ---------------------------------------------------------------------------
// Kernel 2: feature expansion + WMMA GEMM.
// ---------------------------------------------------------------------------
__global__ void __launch_bounds__(THREADS)
kan_gemm(const float* __restrict__ x,
         const _Float16* __restrict__ Wt,
         float* __restrict__ out) {
  __shared__ _Float16 F_lds[M_TILE * K_STRIDE];   // 64 x 168 f16 = 21 KB
  __shared__ _Float16 W_lds[N_TILE * K_STRIDE];   // 64 x 168 f16 = 21 KB

  const int tid  = threadIdx.x;
  const int lane = tid & 31;
  const int wave = tid >> 5;
  const int rb   = blockIdx.x * M_TILE;          // row base of this block
  const int n0   = blockIdx.y * N_TILE;          // col base of this block

  const int mrow = (wave & 1) * 32;              // wave's 32-row strip
  const int ncol = (wave >> 1) * 32;             // wave's 32-col strip

  const int m    = lane & 15;                    // row/col within fragment
  const int kh   = (lane >> 4) * 8;              // K sub-offset per ISA layout

  // acc[s][t]: rows mrow+s*16.., cols ncol+t*16..
  v8f acc00 = v8f{}, acc01 = v8f{}, acc10 = v8f{}, acc11 = v8f{};

  // fragment base pointers within the LDS panels (K offset added per step)
  const _Float16* aB0 = &F_lds[(mrow +  0 + m) * K_STRIDE + kh];
  const _Float16* aB1 = &F_lds[(mrow + 16 + m) * K_STRIDE + kh];
  const _Float16* bB0 = &W_lds[(ncol +  0 + m) * K_STRIDE + kh];
  const _Float16* bB1 = &W_lds[(ncol + 16 + m) * K_STRIDE + kh];

  for (int ic = 0; ic < IN_DIM / I_CHUNK; ++ic) {
    const int ib = ic * I_CHUNK;

    // ---- stage W panel: Wt[n0+oo][ib*FEAT .. +160) -> W_lds[oo][0..160)
    // 64 rows x 20 float4 = 1280 float4 / 128 threads = 10 each.
    {
      const float4* __restrict__ gsrc =
          reinterpret_cast<const float4*>(Wt + (size_t)n0 * KDIM + ib * FEAT);
      float4* ldst = reinterpret_cast<float4*>(W_lds);   // row = 21 float4
#pragma unroll
      for (int j = 0; j < 10; ++j) {
        int q   = tid + j * THREADS;           // 0..1279
        int oo  = q / 20;                      // 20 float4 per 160-half row
        int seg = q % 20;
        ldst[oo * 21 + seg] = gsrc[(size_t)oo * 640 + seg];
      }
      // prefetch next chunk's W panel into cache (global_prefetch_b8)
      if (ic + 1 < IN_DIM / I_CHUNK && tid < 64) {
        __builtin_prefetch(Wt + (size_t)(n0 + tid) * KDIM + (ib + I_CHUNK) * FEAT, 0, 1);
      }
    }

    // ---- compute feature panel: 64 rows x 8 inputs, 4 x-values per thread
#pragma unroll
    for (int e = 0; e < 4; ++e) {
      int p  = tid * 4 + e;                      // 0..511
      int r  = p >> 3;                           // local row 0..63
      int li = p & 7;                            // local input 0..7
      float xv = x[(size_t)(rb + r) * IN_DIM + ib + li];

      // cubic B-spline on uniform grid, step 1/8 over [-1-3h, 1+3h]
      float t = (xv + 1.0f) * 8.0f;
      int   j = (int)floorf(t);
      j = j < 3 ? 3 : (j > 18 ? 18 : j);
      float u  = t - (float)j;
      float um = 1.0f - u;
      float u2 = u * u, u3 = u2 * u;
      const float k6 = 1.0f / 6.0f;
      float B0 = um * um * um * k6;
      float B1 = (3.0f * u3 - 6.0f * u2 + 4.0f) * k6;
      float B2 = (-3.0f * u3 + 3.0f * u2 + 3.0f * u + 1.0f) * k6;
      float B3 = u3 * k6;
      float silu = xv / (1.0f + __expf(-xv));

      float f20[FEAT];
#pragma unroll
      for (int q = 0; q < FEAT; ++q) f20[q] = 0.0f;
      f20[j - 3] = B0; f20[j - 2] = B1; f20[j - 1] = B2; f20[j] = B3;
      f20[FEAT - 1] = silu;

      // pack to 10 dwords, store to LDS (row*336B + li*40B, dword aligned)
      uint32_t* dst = reinterpret_cast<uint32_t*>(&F_lds[r * K_STRIDE + li * FEAT]);
#pragma unroll
      for (int d = 0; d < FEAT / 2; ++d) {
        union { _Float16 h[2]; uint32_t u32; } pk;
        pk.h[0] = (_Float16)f20[2 * d];
        pk.h[1] = (_Float16)f20[2 * d + 1];
        dst[d] = pk.u32;
      }
    }

    __syncthreads();

    // ---- 5 K-steps of 32; software-pipelined fragment loads
    Frag a0, a1, b0, b1;
    a0.h[0] = *(const v8h*)(aB0 +  0); a0.h[1] = *(const v8h*)(aB0 + 16);
    a1.h[0] = *(const v8h*)(aB1 +  0); a1.h[1] = *(const v8h*)(aB1 + 16);
    b0.h[0] = *(const v8h*)(bB0 +  0); b0.h[1] = *(const v8h*)(bB0 + 16);
    b1.h[0] = *(const v8h*)(bB1 +  0); b1.h[1] = *(const v8h*)(bB1 + 16);

#pragma unroll
    for (int ks = 0; ks < 5; ++ks) {
      Frag na0, na1, nb0, nb1;
      if (ks < 4) {
        const int kb = (ks + 1) * 32;
        na0.h[0] = *(const v8h*)(aB0 + kb);      na0.h[1] = *(const v8h*)(aB0 + kb + 16);
        na1.h[0] = *(const v8h*)(aB1 + kb);      na1.h[1] = *(const v8h*)(aB1 + kb + 16);
        nb0.h[0] = *(const v8h*)(bB0 + kb);      nb0.h[1] = *(const v8h*)(bB0 + kb + 16);
        nb1.h[0] = *(const v8h*)(bB1 + kb);      nb1.h[1] = *(const v8h*)(bB1 + kb + 16);
      }
      acc00 = __builtin_amdgcn_wmma_f32_16x16x32_f16(
          false, a0.v, false, b0.v, (short)0, acc00, false, false);
      acc01 = __builtin_amdgcn_wmma_f32_16x16x32_f16(
          false, a0.v, false, b1.v, (short)0, acc01, false, false);
      acc10 = __builtin_amdgcn_wmma_f32_16x16x32_f16(
          false, a1.v, false, b0.v, (short)0, acc10, false, false);
      acc11 = __builtin_amdgcn_wmma_f32_16x16x32_f16(
          false, a1.v, false, b1.v, (short)0, acc11, false, false);
      if (ks < 4) { a0 = na0; a1 = na1; b0 = nb0; b1 = nb1; }
    }

    __syncthreads();
  }

  // ---- store C tiles: VGPR r -> row r (lanes 0-15) / r+8 (lanes 16-31)
  const int rowAdd = (lane >> 4) * 8;
  {
    v8f* accs[4] = { &acc00, &acc01, &acc10, &acc11 };
#pragma unroll
    for (int s = 0; s < 2; ++s) {
#pragma unroll
      for (int t = 0; t < 2; ++t) {
        const v8f& a = *accs[s * 2 + t];
        const int col = n0 + ncol + t * 16 + m;
#pragma unroll
        for (int r = 0; r < 8; ++r) {
          int row = rb + mrow + s * 16 + r + rowAdd;
          out[(size_t)row * OUT_DIM + col] = a[r];
        }
      }
    }
  }
}

// ---------------------------------------------------------------------------
// Host launch
// ---------------------------------------------------------------------------
extern "C" void kernel_launch(void* const* d_in, const int* in_sizes, int n_in,
                              void* d_out, int out_size, void* d_ws, size_t ws_size,
                              hipStream_t stream) {
  const float* x  = (const float*)d_in[0];               // (4096, 256)
  const float* cp = (const float*)d_in[1];               // (256, 256, 19)
  const float* sf = (const float*)d_in[2];               // (256, 256)
  float* out = (float*)d_out;                            // (4096, 256)

  _Float16* Wt = (_Float16*)d_ws;                        // 5120*256*2 = 2.5 MB

  // pack weights: 65536 (i,o) pairs
  kan_pack_weights<<<dim3((IN_DIM * OUT_DIM + 255) / 256), dim3(256), 0, stream>>>(
      cp, sf, Wt);

  // GEMM: 64 M-tiles x 4 N-tiles
  kan_gemm<<<dim3(4096 / M_TILE, OUT_DIM / N_TILE), dim3(THREADS), 0, stream>>>(
      x, Wt, out);
}
